// DFM_34634616275114
// MI455X (gfx1250) — matrix operand order repaired and checked by reading, not compiled, not run on hardware
//
#include <hip/hip_runtime.h>
#include <hip/hip_bf16.h>

typedef __attribute__((ext_vector_type(16))) _Float16 v16h;
typedef __attribute__((ext_vector_type(8)))  float    v8f;

#define NB 8
#define CB 32

// ---------------- helpers ----------------
__device__ __forceinline__ float bilin_clamp(const float* __restrict__ p, int S,
                                             float sy, float sx) {
  float fy = floorf(sy), fx = floorf(sx);
  float wy = sy - fy, wx = sx - fx;
  int y0 = (int)fy, x0 = (int)fx;
  int y1 = y0 + 1, x1 = x0 + 1;
  y0 = min(max(y0, 0), S - 1); y1 = min(max(y1, 0), S - 1);
  x0 = min(max(x0, 0), S - 1); x1 = min(max(x1, 0), S - 1);
  float v00 = p[y0 * S + x0], v01 = p[y0 * S + x1];
  float v10 = p[y1 * S + x0], v11 = p[y1 * S + x1];
  return v00 * (1.f - wy) * (1.f - wx) + v01 * (1.f - wy) * wx
       + v10 * wy * (1.f - wx) + v11 * wy * wx;
}

// ---------------- resize (half-pixel bilinear, antialias=False) into concat buffer ----------------
// in: [8,32,Hi,Hi] -> writes channels [chOff, chOff+32) of cat96 [8,96,32,32]
__global__ void resize_into_cat(const float* __restrict__ in, int Hi,
                                float* __restrict__ cat96, int chOff) {
  int i = blockIdx.x * blockDim.x + threadIdx.x;
  if (i >= NB * CB * 32 * 32) return;
  int x = i & 31, y = (i >> 5) & 31, c = (i >> 10) & 31, n = i >> 15;
  float scale = (float)Hi * (1.0f / 32.0f);
  float sy = (y + 0.5f) * scale - 0.5f;
  float sx = (x + 0.5f) * scale - 0.5f;
  const float* p = in + (size_t)(n * CB + c) * Hi * Hi;
  cat96[((size_t)n * 96 + chOff + c) * 1024 + y * 32 + x] = bilin_clamp(p, Hi, sy, sx);
}

// ---------------- generic 1x1-conv GEMM with WMMA f16 (f32 accumulate) ----------------
// out[n,m,p] = PReLU( sum_k W[m,k]*act[n,k,p] + bias[m] )
// K is a compile-time constant (fully unrolled k-loop); MASK_M only when M%16!=0.
template <int K, bool MASK_M>
__global__ __launch_bounds__(256) void gemm1x1_wmma(
    const float* __restrict__ W, const float* __restrict__ bias,
    const float* __restrict__ slope, const float* __restrict__ act,
    float* __restrict__ out, int M, int P) {
  int lane = threadIdx.x & 31;
  int wave = threadIdx.x >> 5;
  int numPT = P >> 4;
  int ptile = blockIdx.x * 8 + wave;
  if (ptile >= numPT) return;           // wave-uniform: EXEC stays all-ones
  int n = blockIdx.z;
  int mbase = blockIdx.y << 4;
  int pix = (ptile << 4) + (lane & 15);
  int hi = lane >> 4;                   // lane-half
  const float* actn = act + (size_t)n * K * P + pix;
  v8f c = {0.f, 0.f, 0.f, 0.f, 0.f, 0.f, 0.f, 0.f};
  int mg = mbase + (lane & 15);
  int mrow = MASK_M ? min(mg, M - 1) : mg;       // clamp: load stays unconditional
  const float* wrow = W + (size_t)mrow * K;
  int kaBase0 = hi ? 16 : 0;                     // A elems 0..7  -> K = kaBase0 + e
  int kaBase1 = hi ? 24 : 8;                     // A elems 8..15 -> K = kaBase1 + e
  int kbBase = hi ? 16 : 0;                      // B elems e     -> K = kbBase + e
#pragma unroll
  for (int k0 = 0; k0 < K; k0 += 32) {
    v16h a, b;
#pragma unroll
    for (int e = 0; e < 8; ++e) {                // consecutive -> merges to b128 loads
      a[e]     = (_Float16)wrow[k0 + kaBase0 + e];
      a[8 + e] = (_Float16)wrow[k0 + kaBase1 + e];
    }
    if (MASK_M && mg >= M) {
#pragma unroll
      for (int e = 0; e < 16; ++e) a[e] = (_Float16)0.f;   // cndmask, no branch on loads
    }
#pragma unroll
    for (int e = 0; e < 16; ++e)
      b[e] = (_Float16)actn[(size_t)(k0 + kbBase + e) * P];
    c = __builtin_amdgcn_wmma_f32_16x16x32_f16(false, a, false, b, (short)0, c,
                                               false, false);
  }
  float s = slope[0];
  float* outn = out + (size_t)n * M * P + pix;
#pragma unroll
  for (int v = 0; v < 8; ++v) {
    int m = mbase + v + 8 * hi;
    if (!MASK_M || m < M) {
      float x = c[v] + bias[m];
      if (x < 0.0f) x *= s;
      outn[(size_t)m * P] = x;
    }
  }
}

// ---------------- GAP over 32x32 ----------------
__global__ void gap_kernel(const float* __restrict__ fused, float* __restrict__ gapv) {
  __shared__ float sm[256];
  const float* p = fused + (size_t)blockIdx.x * 1024;
  float s = 0.f;
  for (int i = threadIdx.x; i < 1024; i += 256) s += p[i];
  sm[threadIdx.x] = s;
  __syncthreads();
  for (int w = 128; w > 0; w >>= 1) {
    if (threadIdx.x < w) sm[threadIdx.x] += sm[threadIdx.x + w];
    __syncthreads();
  }
  if (threadIdx.x == 0) gapv[blockIdx.x] = sm[0] * (1.0f / 1024.0f);
}

// ---------------- pixel_shuffle(tch) [+ tsp] -> softmax over 9 ----------------
// tch: [8, 9*r*r, Sin*Sin], tsp: [8,9,H*H] or null, kout: [8,9,H,H]
__global__ void kmap_softmax(const float* __restrict__ tch, const float* __restrict__ tsp,
                             float* __restrict__ kout, int r, int H) {
  int i = blockIdx.x * blockDim.x + threadIdx.x;
  int HH = H * H;
  if (i >= NB * HH) return;
  int x = i % H, y = (i / H) % H, n = i / HH;
  int Sin = H / r, rr = r * r;
  int off = (y % r) * r + (x % r);
  int ys = y / r, xs = x / r;
  float v[9], mx = -1e30f;
#pragma unroll
  for (int k = 0; k < 9; ++k) {
    float t = tch[(((size_t)n * 9 + k) * rr + off) * (Sin * Sin) + ys * Sin + xs];
    if (tsp) t += tsp[((size_t)n * 9 + k) * HH + y * H + x];
    v[k] = t;
    mx = fmaxf(mx, t);
  }
  float s = 0.f;
#pragma unroll
  for (int k = 0; k < 9; ++k) { v[k] = expf(v[k] - mx); s += v[k]; }
  float inv = 1.0f / s;
#pragma unroll
  for (int k = 0; k < 9; ++k)
    kout[((size_t)n * 9 + k) * HH + y * H + x] = v[k] * inv;
}

// ---------------- pixel_shuffle(t) + fea -> g ----------------
// t: [8, 32*r*r, Sin*Sin], fea/g: [8,32,H,H]
__global__ void psadd_kernel(const float* __restrict__ t, const float* __restrict__ fea,
                             float* __restrict__ g, int r, int H) {
  int i = blockIdx.x * blockDim.x + threadIdx.x;
  int HH = H * H;
  if (i >= NB * CB * HH) return;
  int x = i % H, y = (i / H) % H, c = (i / HH) % CB, n = i / (CB * HH);
  int Sin = H / r, rr = r * r;
  int off = (y % r) * r + (x % r);
  float v = t[(((size_t)n * CB + c) * rr + off) * (Sin * Sin) + (y / r) * Sin + (x / r)];
  g[i] = v + fea[i];
}

// ---------------- adaptive_pool3 of zero-padded g (pool/conv commuted) ----------------
// g: [8,32,S,S]; pooled: [8*32, 9] mean over padded (S+2) bins
__global__ void pool3_kernel(const float* __restrict__ g, float* __restrict__ pooled, int S) {
  __shared__ float sm[64];
  int b = blockIdx.x % 9;
  int nc = blockIdx.x / 9;
  int i = b / 3, j = b % 3;
  int Sp = S + 2;
  int r0 = (i * Sp) / 3, r1 = ((i + 1) * Sp + 2) / 3;
  int c0 = (j * Sp) / 3, c1 = ((j + 1) * Sp + 2) / 3;
  int vr0 = max(r0 - 1, 0), vr1 = min(r1 - 1, S);
  int vc0 = max(c0 - 1, 0), vc1 = min(c1 - 1, S);
  int ncl = vc1 - vc0, tot = (vr1 - vr0) * ncl;
  const float* p = g + (size_t)nc * S * S;
  float sum = 0.f;
  for (int t = threadIdx.x; t < tot; t += 64) {
    int rr = vr0 + t / ncl, cc = vc0 + t % ncl;
    sum += p[rr * S + cc];
  }
  sm[threadIdx.x] = sum;
  __syncthreads();
  for (int w = 32; w > 0; w >>= 1) {
    if (threadIdx.x < w) sm[threadIdx.x] += sm[threadIdx.x + w];
    __syncthreads();
  }
  if (threadIdx.x == 0)
    pooled[(size_t)nc * 9 + b] = sm[0] / (float)((r1 - r0) * (c1 - c0));
}

// ---------------- gkc tail: K = prelu(W2*(W1*pooled+b1)+b2) ----------------
// pooled,Kout: [8,32,9]; grid = 8*9 blocks of 32 threads
__global__ void gkc_kernel(const float* __restrict__ pooled,
                           const float* __restrict__ W1, const float* __restrict__ b1,
                           const float* __restrict__ W2, const float* __restrict__ b2,
                           const float* __restrict__ slope, float* __restrict__ Kout) {
  __shared__ float y1[32];
  int n = blockIdx.x / 9, b = blockIdx.x % 9;
  int co = threadIdx.x;
  float s = 0.f;
#pragma unroll 8
  for (int c = 0; c < 32; ++c) s += W1[co * 32 + c] * pooled[((size_t)n * 32 + c) * 9 + b];
  y1[co] = s + b1[co];
  __syncthreads();
  float t = 0.f;
#pragma unroll 8
  for (int c = 0; c < 32; ++c) t += W2[co * 32 + c] * y1[c];
  t += b2[co];
  float a = slope[0];
  if (t < 0.f) t *= a;
  Kout[((size_t)n * 32 + co) * 9 + b] = t;
}

// ---------------- 3x3 patch combine: softmax-weighted + per-(n,c) kernel ----------------
__global__ void patch_kernel(const float* __restrict__ fea, const float* __restrict__ gapv,
                             const float* __restrict__ kmap, const float* __restrict__ Kmat,
                             float* __restrict__ out_new, float* __restrict__ out_ch, int H) {
  int i = blockIdx.x * blockDim.x + threadIdx.x;
  int HH = H * H;
  if (i >= NB * CB * HH) return;
  int x = i % H, y = (i / H) % H, c = (i / HH) % CB, n = i / (CB * HH);
  float g = gapv[n * CB + c];
  const float* f = fea + ((size_t)n * CB + c) * HH;
  const float* km = kmap + (size_t)n * 9 * HH + y * H + x;
  const float* Kc = Kmat + ((size_t)n * CB + c) * 9;
  float accN = 0.f, accC = 0.f;
#pragma unroll
  for (int dh = 0; dh < 3; ++dh)
#pragma unroll
    for (int dw = 0; dw < 3; ++dw) {
      int yy = y + dh - 1, xx = x + dw - 1, k = dh * 3 + dw;
      float pv = (yy >= 0 && yy < H && xx >= 0 && xx < H) ? (f[yy * H + xx] + g) : 0.f;
      accN += pv * km[(size_t)k * HH];
      accC += pv * Kc[k];
    }
  out_new[i] = accN;
  out_ch[i] = accC;
}

// ---------------- final fuse GEMM: cat of 6 groups, 4 upsampled on the fly ----------------
// G (channel group) is a true template constant -> all branches fold at compile time.
template <int G>
__device__ __forceinline__ float load_cat_g(int n, int c, int y, int x,
    const float* __restrict__ hn, const float* __restrict__ chh,
    const float* __restrict__ mn, const float* __restrict__ chm,
    const float* __restrict__ ln, const float* __restrict__ chl) {
  if (G == 4) return ln[((size_t)n * CB + c) * 16384 + (y << 7) + x];
  if (G == 5) return chl[((size_t)n * CB + c) * 16384 + (y << 7) + x];
  const int S = (G < 2) ? 32 : 64;
  const float* src = (G == 0) ? hn : (G == 1) ? chh : (G == 2) ? mn : chm;
  float scale = (float)S * (1.0f / 128.0f);
  float sy = (y + 0.5f) * scale - 0.5f;
  float sx = (x + 0.5f) * scale - 0.5f;
  return bilin_clamp(src + ((size_t)n * CB + c) * S * S, S, sy, sx);
}

// one K-step (32 channels = one cat group) of the final GEMM
template <int G>
__device__ __forceinline__ void fuse_step(v8f& c, const float* __restrict__ wrow,
    int kaBase0, int kaBase1, int coff, int n, int y, int x,
    const float* __restrict__ hn, const float* __restrict__ chh,
    const float* __restrict__ mn, const float* __restrict__ chm,
    const float* __restrict__ ln, const float* __restrict__ chl) {
  const int k0 = G * 32;
  v16h a, b;
#pragma unroll
  for (int e = 0; e < 8; ++e) {
    a[e]     = (_Float16)wrow[k0 + kaBase0 + e];
    a[8 + e] = (_Float16)wrow[k0 + kaBase1 + e];
  }
#pragma unroll
  for (int e = 0; e < 16; ++e)
    b[e] = (_Float16)load_cat_g<G>(n, coff + e, y, x, hn, chh, mn, chm, ln, chl);
  c = __builtin_amdgcn_wmma_f32_16x16x32_f16(false, a, false, b, (short)0, c,
                                             false, false);
}

__global__ __launch_bounds__(256) void gemm_final_wmma(
    const float* __restrict__ W, const float* __restrict__ slope,
    const float* __restrict__ hn, const float* __restrict__ chh,
    const float* __restrict__ mn, const float* __restrict__ chm,
    const float* __restrict__ ln, const float* __restrict__ chl,
    float* __restrict__ out) {
  const int P = 16384, K = 192, M = 32;
  int lane = threadIdx.x & 31;
  int wave = threadIdx.x >> 5;
  int ptile = blockIdx.x * 8 + wave;
  if (ptile >= (P >> 4)) return;
  int n = blockIdx.z;
  int mbase = blockIdx.y << 4;
  int pix = (ptile << 4) + (lane & 15);
  int y = pix >> 7, x = pix & 127;
  int hi = lane >> 4;
  const float* wrow = W + (size_t)(mbase + (lane & 15)) * K;
  int kaBase0 = hi ? 16 : 0;
  int kaBase1 = hi ? 24 : 8;
  int coff = hi ? 16 : 0;   // within-group channel offset for B elements
  v8f c = {0.f, 0.f, 0.f, 0.f, 0.f, 0.f, 0.f, 0.f};
  fuse_step<0>(c, wrow, kaBase0, kaBase1, coff, n, y, x, hn, chh, mn, chm, ln, chl);
  fuse_step<1>(c, wrow, kaBase0, kaBase1, coff, n, y, x, hn, chh, mn, chm, ln, chl);
  fuse_step<2>(c, wrow, kaBase0, kaBase1, coff, n, y, x, hn, chh, mn, chm, ln, chl);
  fuse_step<3>(c, wrow, kaBase0, kaBase1, coff, n, y, x, hn, chh, mn, chm, ln, chl);
  fuse_step<4>(c, wrow, kaBase0, kaBase1, coff, n, y, x, hn, chh, mn, chm, ln, chl);
  fuse_step<5>(c, wrow, kaBase0, kaBase1, coff, n, y, x, hn, chh, mn, chm, ln, chl);
  float s = slope[0];
#pragma unroll
  for (int v = 0; v < 8; ++v) {
    int m = mbase + v + 8 * hi;
    float val = c[v];
    if (val < 0.f) val *= s;
    out[((size_t)n * M + m) * P + pix] = val;
  }
}

// ---------------- launcher ----------------
static inline dim3 ewgrid(size_t total) { return dim3((unsigned)((total + 255) / 256)); }

extern "C" void kernel_launch(void* const* d_in, const int* in_sizes, int n_in,
                              void* d_out, int out_size, void* d_ws, size_t ws_size,
                              hipStream_t stream) {
  (void)in_sizes; (void)n_in; (void)out_size; (void)ws_size;
  const float* fea_low  = (const float*)d_in[0];
  const float* fea_mid  = (const float*)d_in[1];
  const float* fea_high = (const float*)d_in[2];
  const float* fuse_high_W = (const float*)d_in[3];
  const float* fuse_high_b = (const float*)d_in[4];
  const float* fuse_high_a = (const float*)d_in[5];
  const float* fuse_W = (const float*)d_in[6];
  const float* fuse_a = (const float*)d_in[7];
  const float* ch_high_W = (const float*)d_in[8];
  const float* ch_high_b = (const float*)d_in[9];
  const float* ch_high_a = (const float*)d_in[10];
  const float* ch_mid_W = (const float*)d_in[11];
  const float* ch_mid_b = (const float*)d_in[12];
  const float* ch_mid_a = (const float*)d_in[13];
  const float* ch_low_W = (const float*)d_in[14];
  const float* ch_low_b = (const float*)d_in[15];
  const float* ch_low_a = (const float*)d_in[16];
  const float* low_sp_W = (const float*)d_in[17];
  const float* low_sp_b = (const float*)d_in[18];
  const float* low_sp_a = (const float*)d_in[19];
  const float* mid_sp_W = (const float*)d_in[20];
  const float* mid_sp_b = (const float*)d_in[21];
  const float* mid_sp_a = (const float*)d_in[22];
  const float* resize_low_W = (const float*)d_in[23];
  const float* resize_low_b = (const float*)d_in[24];
  const float* resize_low_a = (const float*)d_in[25];
  const float* resize_mid_W = (const float*)d_in[26];
  const float* resize_mid_b = (const float*)d_in[27];
  const float* resize_mid_a = (const float*)d_in[28];
  const float* gkc_low_W1 = (const float*)d_in[29];
  const float* gkc_low_b1 = (const float*)d_in[30];
  const float* gkc_low_W2 = (const float*)d_in[31];
  const float* gkc_low_b2 = (const float*)d_in[32];
  const float* gkc_low_a  = (const float*)d_in[33];
  const float* gkc_mid_W1 = (const float*)d_in[34];
  const float* gkc_mid_b1 = (const float*)d_in[35];
  const float* gkc_mid_W2 = (const float*)d_in[36];
  const float* gkc_mid_b2 = (const float*)d_in[37];
  const float* gkc_mid_a  = (const float*)d_in[38];
  const float* gkc_high_W1 = (const float*)d_in[39];
  const float* gkc_high_b1 = (const float*)d_in[40];
  const float* gkc_high_W2 = (const float*)d_in[41];
  const float* gkc_high_b2 = (const float*)d_in[42];
  const float* gkc_high_a  = (const float*)d_in[43];
  float* out = (float*)d_out;

  float* ws = (float*)d_ws;
  size_t o = 0;
  float* cat96   = ws + o; o += (size_t)NB * 96 * 1024;
  float* fused   = ws + o; o += (size_t)NB * 32 * 1024;
  float* gapv    = ws + o; o += (size_t)NB * 32;
  float* t_chlow = ws + o; o += (size_t)NB * 144 * 1024;
  float* t_lowsp = ws + o; o += (size_t)NB * 9 * 16384;
  float* t_chmid = ws + o; o += (size_t)NB * 36 * 1024;
  float* t_midsp = ws + o; o += (size_t)NB * 9 * 4096;
  float* t_chhigh= ws + o; o += (size_t)NB * 9 * 1024;
  float* k_low   = ws + o; o += (size_t)NB * 9 * 16384;
  float* k_mid   = ws + o; o += (size_t)NB * 9 * 4096;
  float* k_high  = ws + o; o += (size_t)NB * 9 * 1024;
  float* t_rlow  = ws + o; o += (size_t)NB * 512 * 1024;
  float* t_rmid  = ws + o; o += (size_t)NB * 128 * 1024;
  float* g_low   = ws + o; o += (size_t)NB * 32 * 16384;
  float* g_mid   = ws + o; o += (size_t)NB * 32 * 4096;
  float* pooled_low  = ws + o; o += (size_t)NB * 32 * 9;
  float* pooled_mid  = ws + o; o += (size_t)NB * 32 * 9;
  float* pooled_high = ws + o; o += (size_t)NB * 32 * 9;
  float* K_low  = ws + o; o += (size_t)NB * 32 * 9;
  float* K_mid  = ws + o; o += (size_t)NB * 32 * 9;
  float* K_high = ws + o; o += (size_t)NB * 32 * 9;
  float* low_new = ws + o; o += (size_t)NB * 32 * 16384;
  float* ch_lowb = ws + o; o += (size_t)NB * 32 * 16384;
  float* mid_new = ws + o; o += (size_t)NB * 32 * 4096;
  float* ch_midb = ws + o; o += (size_t)NB * 32 * 4096;
  float* high_new= ws + o; o += (size_t)NB * 32 * 1024;
  float* ch_highb= ws + o; o += (size_t)NB * 32 * 1024;

  // 1) build 96-channel concat at 32x32
  hipLaunchKernelGGL(resize_into_cat, ewgrid((size_t)NB*CB*1024), dim3(256), 0, stream, fea_high, 32,  cat96, 0);
  hipLaunchKernelGGL(resize_into_cat, ewgrid((size_t)NB*CB*1024), dim3(256), 0, stream, fea_mid,  64,  cat96, 32);
  hipLaunchKernelGGL(resize_into_cat, ewgrid((size_t)NB*CB*1024), dim3(256), 0, stream, fea_low,  128, cat96, 64);

  // 2) fused = prelu(conv1x1(cat96))
  hipLaunchKernelGGL((gemm1x1_wmma<96, false>), dim3(8, 2, NB), dim3(256), 0, stream,
                     fuse_high_W, fuse_high_b, fuse_high_a, cat96, fused, 32, 1024);
  // 3) GAP
  hipLaunchKernelGGL(gap_kernel, dim3(NB * CB), dim3(256), 0, stream, fused, gapv);

  // 4) channel-path and spatial-path convs
  hipLaunchKernelGGL((gemm1x1_wmma<32, false>), dim3(8, 9, NB), dim3(256), 0, stream,
                     ch_low_W, ch_low_b, ch_low_a, fused, t_chlow, 144, 1024);
  hipLaunchKernelGGL((gemm1x1_wmma<32, true>), dim3(128, 1, NB), dim3(256), 0, stream,
                     low_sp_W, low_sp_b, low_sp_a, fea_low, t_lowsp, 9, 16384);
  hipLaunchKernelGGL((gemm1x1_wmma<32, true>), dim3(8, 3, NB), dim3(256), 0, stream,
                     ch_mid_W, ch_mid_b, ch_mid_a, fused, t_chmid, 36, 1024);
  hipLaunchKernelGGL((gemm1x1_wmma<32, true>), dim3(32, 1, NB), dim3(256), 0, stream,
                     mid_sp_W, mid_sp_b, mid_sp_a, fea_mid, t_midsp, 9, 4096);
  hipLaunchKernelGGL((gemm1x1_wmma<32, true>), dim3(8, 1, NB), dim3(256), 0, stream,
                     ch_high_W, ch_high_b, ch_high_a, fused, t_chhigh, 9, 1024);

  // 5) softmax k-maps
  hipLaunchKernelGGL(kmap_softmax, ewgrid((size_t)NB*16384), dim3(256), 0, stream, t_chlow,  t_lowsp, k_low, 4, 128);
  hipLaunchKernelGGL(kmap_softmax, ewgrid((size_t)NB*4096),  dim3(256), 0, stream, t_chmid,  t_midsp, k_mid, 2, 64);
  hipLaunchKernelGGL(kmap_softmax, ewgrid((size_t)NB*1024),  dim3(256), 0, stream, t_chhigh, (const float*)nullptr, k_high, 1, 32);

  // 6) resize-channel convs + pixel-shuffle add -> gkc inputs
  hipLaunchKernelGGL((gemm1x1_wmma<32, false>), dim3(8, 32, NB), dim3(256), 0, stream,
                     resize_low_W, resize_low_b, resize_low_a, fused, t_rlow, 512, 1024);
  hipLaunchKernelGGL((gemm1x1_wmma<32, false>), dim3(8, 8, NB), dim3(256), 0, stream,
                     resize_mid_W, resize_mid_b, resize_mid_a, fused, t_rmid, 128, 1024);
  hipLaunchKernelGGL(psadd_kernel, ewgrid((size_t)NB*CB*16384), dim3(256), 0, stream, t_rlow, fea_low, g_low, 4, 128);
  hipLaunchKernelGGL(psadd_kernel, ewgrid((size_t)NB*CB*4096),  dim3(256), 0, stream, t_rmid, fea_mid, g_mid, 2, 64);

  // 7) adaptive pool3 (pool commuted before gkc's 1x1 conv) + gkc tail
  hipLaunchKernelGGL(pool3_kernel, dim3(NB * CB * 9), dim3(64), 0, stream, g_low, pooled_low, 128);
  hipLaunchKernelGGL(pool3_kernel, dim3(NB * CB * 9), dim3(64), 0, stream, g_mid, pooled_mid, 64);
  hipLaunchKernelGGL(pool3_kernel, dim3(NB * CB * 9), dim3(64), 0, stream, fused, pooled_high, 32);
  hipLaunchKernelGGL(gkc_kernel, dim3(NB * 9), dim3(32), 0, stream,
                     pooled_low, gkc_low_W1, gkc_low_b1, gkc_low_W2, gkc_low_b2, gkc_low_a, K_low);
  hipLaunchKernelGGL(gkc_kernel, dim3(NB * 9), dim3(32), 0, stream,
                     pooled_mid, gkc_mid_W1, gkc_mid_b1, gkc_mid_W2, gkc_mid_b2, gkc_mid_a, K_mid);
  hipLaunchKernelGGL(gkc_kernel, dim3(NB * 9), dim3(32), 0, stream,
                     pooled_high, gkc_high_W1, gkc_high_b1, gkc_high_W2, gkc_high_b2, gkc_high_a, K_high);

  // 8) patch combines (both weighted outputs per scale)
  hipLaunchKernelGGL(patch_kernel, ewgrid((size_t)NB*CB*16384), dim3(256), 0, stream,
                     fea_low, gapv, k_low, K_low, low_new, ch_lowb, 128);
  hipLaunchKernelGGL(patch_kernel, ewgrid((size_t)NB*CB*4096), dim3(256), 0, stream,
                     fea_mid, gapv, k_mid, K_mid, mid_new, ch_midb, 64);
  hipLaunchKernelGGL(patch_kernel, ewgrid((size_t)NB*CB*1024), dim3(256), 0, stream,
                     fea_high, gapv, k_high, K_high, high_new, ch_highb, 32);

  // 9) final fuse conv over synthesized 192-channel cat
  hipLaunchKernelGGL(gemm_final_wmma, dim3(128, 2, NB), dim3(256), 0, stream,
                     fuse_W, fuse_a, high_new, ch_highb, mid_new, ch_midb, low_new, ch_lowb, out);
}
